// AttentiveFPModel_42958262895199
// MI455X (gfx1250) — matrix operand order repaired and checked by reading, not compiled, not run on hardware
//
#include <hip/hip_runtime.h>
#include <hip/hip_bf16.h>
#include <math.h>

// ---------------- problem constants ----------------
constexpr int Nn  = 100000;   // nodes (mult of 16)
constexpr int Ee  = 400000;   // edges (mult of 16)
constexpr int Ff  = 143;      // input feature dim
constexpr int Edw = 6;        // edge feature dim
constexpr int Hh  = 128;      // hidden
constexpr int Gg  = 4096;     // graphs (mult of 16)
constexpr int KFp = 160;      // 143 padded to mult of 32
constexpr int KCp = 160;      // 134 padded to mult of 32

typedef __attribute__((ext_vector_type(16))) __bf16 v16bf;
typedef __attribute__((ext_vector_type(8)))  __bf16 v8bf;
typedef __attribute__((ext_vector_type(8)))  float  v8f;

__device__ __forceinline__ float leakyf(float x){ return x >= 0.f ? x : 0.01f * x; }
__device__ __forceinline__ float sigmf(float x){ return 1.f / (1.f + __expf(-x)); }
__device__ __forceinline__ unsigned fkey(float f){
    unsigned b = __float_as_uint(f);
    return (b & 0x80000000u) ? ~b : (b | 0x80000000u);
}
__device__ __forceinline__ float unkey(unsigned k){
    return (k & 0x80000000u) ? __uint_as_float(k & 0x7fffffffu) : __uint_as_float(~k);
}
__device__ __forceinline__ v8f v8fz(){ v8f z = {0.f,0.f,0.f,0.f,0.f,0.f,0.f,0.f}; return z; }

// Assemble A fragment (16x32 bf16, MxK) per CDNA5 wave32 layout from two contiguous
// 8-half chunks: VGPRs 0-3 hold K = kb+8*half+{0..7}, VGPRs 4-7 hold K = kb+16+8*half+{0..7}
__device__ __forceinline__ v16bf make_a(const v8bf lo, const v8bf hi){
    v16bf a;
#pragma unroll
    for (int i = 0; i < 8; i++){ a[i] = lo[i]; a[8 + i] = hi[i]; }
    return a;
}

// ---------------- generic WMMA GEMM: Y[M,Nout] = act(A[M,KP] @ W[Nout,KP]^T + bias) --------------
// A bf16 row-major, W bf16 row-major [Nout,KP] (== B column strips contiguous in K).
// One wave computes a 16x64 tile (4 wmma accumulators). KP compile-time => K loop fully unrolled,
// all loads use immediate offsets from loop-invariant base pointers, batched before the wmma chain.
// ACT: 0=none 1=leaky 2=relu
template <int ACT, int KP>
__global__ void k_gemm(const __bf16* __restrict__ A, const __bf16* __restrict__ W,
                       const float* __restrict__ bias, float* __restrict__ Y,
                       int M, int Nout)
{
    const int lane = threadIdx.x & 31;
    const int half = lane >> 4, li = lane & 15;
    const int wid  = blockIdx.x * (blockDim.x >> 5) + (threadIdx.x >> 5);
    const int nw   = gridDim.x * (blockDim.x >> 5);
    const int rtiles = M >> 4;
    const int cgrps  = Nout >> 6;
    const long total = (long)rtiles * cgrps;
    for (long t = wid; t < total; t += nw) {
        const int rt = (int)(t / cgrps), cg = (int)(t % cgrps);
        const int row0 = rt << 4, col0 = cg << 6;
        v8f acc[4];
#pragma unroll
        for (int i = 0; i < 4; i++) acc[i] = v8fz();
        const __bf16* pa = A + (long)(row0 + li) * KP + 8 * half;
        const __bf16* pb[4];
#pragma unroll
        for (int ct = 0; ct < 4; ct++)
            pb[ct] = W + (long)(col0 + ct * 16 + li) * KP + 8 * half;
#pragma unroll
        for (int kb = 0; kb < KP; kb += 32) {
            // batch all loads for this K step, then run the independent wmma chain
            const v8bf alo = *(const v8bf*)(pa + kb);
            const v8bf ahi = *(const v8bf*)(pa + kb + 16);
            v16bf b[4];
#pragma unroll
            for (int ct = 0; ct < 4; ct++) b[ct] = make_a(*(const v8bf*)(pb[ct] + kb), *(const v8bf*)(pb[ct] + kb + 16));
            const v16bf a = make_a(alo, ahi);
#pragma unroll
            for (int ct = 0; ct < 4; ct++)
                acc[ct] = __builtin_amdgcn_wmma_f32_16x16x32_bf16(
                    false, a, false, b[ct], (short)0, acc[ct], false, false);
        }
#pragma unroll
        for (int ct = 0; ct < 4; ct++) {
            const int col = col0 + ct * 16 + li;
            const float bv = bias ? bias[col] : 0.f;
#pragma unroll
            for (int v = 0; v < 8; v++) {
                const long row = row0 + v + 8 * half;
                float x = acc[ct][v] + bv;
                if (ACT == 1) x = leakyf(x);
                else if (ACT == 2) x = x > 0.f ? x : 0.f;
                Y[row * (long)Nout + col] = x;
            }
        }
    }
}

// ---------------- fused GATEConv edge projection + attention dot ----------------
// out[row] = sum_c leaky((A@W^T)[row,c]) * attl[c], Nout fixed at 128 (8 accumulators).
// Avoids materializing the [E,128] intermediate (205 MB).
template <int KP>
__global__ void k_gate_gemm_rowdot(const __bf16* __restrict__ A, const __bf16* __restrict__ W,
                                   const float* __restrict__ attl, float* __restrict__ out,
                                   int M)
{
    const int lane = threadIdx.x & 31;
    const int half = lane >> 4, li = lane & 15;
    const int wid  = blockIdx.x * (blockDim.x >> 5) + (threadIdx.x >> 5);
    const int nw   = gridDim.x * (blockDim.x >> 5);
    const int rtiles = M >> 4;
    for (int rt = wid; rt < rtiles; rt += nw) {
        const int row0 = rt << 4;
        v8f acc[8];
#pragma unroll
        for (int i = 0; i < 8; i++) acc[i] = v8fz();
        const __bf16* pa = A + (long)(row0 + li) * KP + 8 * half;
        const __bf16* pb[8];
#pragma unroll
        for (int ct = 0; ct < 8; ct++)
            pb[ct] = W + (long)(ct * 16 + li) * KP + 8 * half;
#pragma unroll
        for (int kb = 0; kb < KP; kb += 32) {
            const v8bf alo = *(const v8bf*)(pa + kb);
            const v8bf ahi = *(const v8bf*)(pa + kb + 16);
            v16bf b[8];
#pragma unroll
            for (int ct = 0; ct < 8; ct++) b[ct] = make_a(*(const v8bf*)(pb[ct] + kb), *(const v8bf*)(pb[ct] + kb + 16));
            const v16bf a = make_a(alo, ahi);
#pragma unroll
            for (int ct = 0; ct < 8; ct++)
                acc[ct] = __builtin_amdgcn_wmma_f32_16x16x32_bf16(
                    false, a, false, b[ct], (short)0, acc[ct], false, false);
        }
        float part[8];
#pragma unroll
        for (int v = 0; v < 8; v++) part[v] = 0.f;
#pragma unroll
        for (int ct = 0; ct < 8; ct++) {
            const float av = attl[ct * 16 + li];
#pragma unroll
            for (int v = 0; v < 8; v++) part[v] += leakyf(acc[ct][v]) * av;
        }
        // reduce across the 16 lanes sharing the same half (cols)
#pragma unroll
        for (int v = 0; v < 8; v++) {
            part[v] += __shfl_xor(part[v], 1, 32);
            part[v] += __shfl_xor(part[v], 2, 32);
            part[v] += __shfl_xor(part[v], 4, 32);
            part[v] += __shfl_xor(part[v], 8, 32);
        }
        if (li == 0) {
#pragma unroll
            for (int v = 0; v < 8; v++) out[row0 + 8 * half + v] = part[v];
        }
    }
}

// ---------------- elementwise / utility kernels ----------------
__global__ void k_cvt_pad(const float* __restrict__ src, __bf16* __restrict__ dst,
                          long rows, int cols, int colsp)
{
    const long total = rows * (long)colsp;
    for (long i = blockIdx.x * (long)blockDim.x + threadIdx.x; i < total;
         i += (long)gridDim.x * blockDim.x) {
        const long r = i / colsp;
        const int  c = (int)(i - r * colsp);
        const float v = (c < cols) ? src[r * (long)cols + c] : 0.f;
        dst[i] = (__bf16)v;
    }
}

__global__ void k_fill_u32(unsigned* __restrict__ p, unsigned v, long n)
{
    for (long i = blockIdx.x * (long)blockDim.x + threadIdx.x; i < n;
         i += (long)gridDim.x * blockDim.x) p[i] = v;
}

__global__ void k_rows_bias(float* __restrict__ dst, const float* __restrict__ bias, long rows)
{
    const long total = rows * Hh;
    for (long i = blockIdx.x * (long)blockDim.x + threadIdx.x; i < total;
         i += (long)gridDim.x * blockDim.x) dst[i] = bias[i & (Hh - 1)];
}

// build Ecat[e, 0:160] = { x0bf[src[e], 0:128], bf16(edge_attr[e,0:6]), zeros }
__global__ void k_ecat(const __bf16* __restrict__ x0bf, const float* __restrict__ eattr,
                       const int* __restrict__ src, __bf16* __restrict__ ecat)
{
    const long total = (long)Ee * KCp;
    for (long i = blockIdx.x * (long)blockDim.x + threadIdx.x; i < total;
         i += (long)gridDim.x * blockDim.x) {
        const long e = i / KCp;
        const int  c = (int)(i - e * KCp);
        __bf16 v;
        if (c < Hh)             v = x0bf[(long)src[e] * Hh + c];
        else if (c < Hh + Edw)  v = (__bf16)eattr[e * Edw + (c - Hh)];
        else                    v = (__bf16)0.f;
        ecat[i] = v;
    }
}

// per-row dot with one or two H-length vectors; one wave per row
__global__ void k_rowdot2(const float* __restrict__ X, const float* __restrict__ w1,
                          const float* __restrict__ w2, float* __restrict__ o1,
                          float* __restrict__ o2, int M)
{
    const int lane = threadIdx.x & 31;
    int row = blockIdx.x * (blockDim.x >> 5) + (threadIdx.x >> 5);
    const int st = gridDim.x * (blockDim.x >> 5);
    for (; row < M; row += st) {
        const float4 v = ((const float4*)(X + (long)row * Hh))[lane];
        const float4 a = ((const float4*)w1)[lane];
        float s1 = v.x * a.x + v.y * a.y + v.z * a.z + v.w * a.w;
        float s2 = 0.f;
        if (w2) {
            const float4 b = ((const float4*)w2)[lane];
            s2 = v.x * b.x + v.y * b.y + v.z * b.z + v.w * b.w;
        }
#pragma unroll
        for (int off = 16; off; off >>= 1) {
            s1 += __shfl_xor(s1, off, 32);
            s2 += __shfl_xor(s2, off, 32);
        }
        if (lane == 0) { o1[row] = s1; if (o2) o2[row] = s2; }
    }
}

// alpha[e] = leaky((sidx? p[sidx[e]] : p[e]) + q[didx[e]]); fused segment-max (uint key trick)
__global__ void k_edge_alpha(const float* __restrict__ p, const int* __restrict__ sidx,
                             const float* __restrict__ q, const int* __restrict__ didx,
                             float* __restrict__ alpha, unsigned* __restrict__ mkey, int n)
{
    for (int e = blockIdx.x * blockDim.x + threadIdx.x; e < n; e += gridDim.x * blockDim.x) {
        const int d = didx[e];
        const float pv = sidx ? p[sidx[e]] : p[e];
        const float a = leakyf(pv + q[d]);
        alpha[e] = a;
        atomicMax(&mkey[d], fkey(a));
    }
}

__global__ void k_expsum(const float* __restrict__ alpha, const int* __restrict__ didx,
                         const unsigned* __restrict__ mkey, float* __restrict__ ev,
                         float* __restrict__ s, int n)
{
    for (int e = blockIdx.x * blockDim.x + threadIdx.x; e < n; e += gridDim.x * blockDim.x) {
        const int d = didx[e];
        const float x = __expf(alpha[e] - unkey(mkey[d]));
        ev[e] = x;
        atomicAdd(&s[d], x);
    }
}

// agg[didx[e], :] += X[row, :] * (ev? ev[e]/(s[d]+1e-16) : 1); one wave per item, 4 floats/lane
__global__ void k_scatter(const float* __restrict__ X, const int* __restrict__ ridx,
                          const int* __restrict__ didx, const float* __restrict__ ev,
                          const float* __restrict__ s, float* __restrict__ agg, int n)
{
    const int lane = threadIdx.x & 31;
    int e = blockIdx.x * (blockDim.x >> 5) + (threadIdx.x >> 5);
    const int st = gridDim.x * (blockDim.x >> 5);
    for (; e < n; e += st) {
        const int r = ridx ? ridx[e] : e;
        const int d = didx[e];
        const float w = ev ? ev[e] / (s[d] + 1e-16f) : 1.f;
        const float4 v = ((const float4*)(X + (long)r * Hh))[lane];
        float* a = agg + (long)d * Hh + lane * 4;
        atomicAdd(a + 0, v.x * w);
        atomicAdd(a + 1, v.y * w);
        atomicAdd(a + 2, v.z * w);
        atomicAdd(a + 3, v.w * w);
    }
}

__global__ void k_elu_bf16(const float* __restrict__ s, __bf16* __restrict__ d, long n)
{
    for (long i = blockIdx.x * (long)blockDim.x + threadIdx.x; i < n;
         i += (long)gridDim.x * blockDim.x) {
        const float x = s[i];
        d[i] = (__bf16)(x > 0.f ? x : __expf(x) - 1.f);
    }
}

__global__ void k_relu_fb(const float* __restrict__ s, float* __restrict__ d,
                          __bf16* __restrict__ dbf, long n)
{
    for (long i = blockIdx.x * (long)blockDim.x + threadIdx.x; i < n;
         i += (long)gridDim.x * blockDim.x) {
        const float v = s[i] > 0.f ? s[i] : 0.f;
        d[i] = v; dbf[i] = (__bf16)v;
    }
}

// GRU gates (order r,z,n) fused with final relu; h updated in place, bf16 copy emitted
__global__ void k_gru(const float* __restrict__ gi, const float* __restrict__ gh,
                      float* __restrict__ h, __bf16* __restrict__ hbf, long M)
{
    const long total = M * Hh;
    for (long i = blockIdx.x * (long)blockDim.x + threadIdx.x; i < total;
         i += (long)gridDim.x * blockDim.x) {
        const long nidx = i >> 7;
        const int  c = (int)(i & (Hh - 1));
        const float* a = gi + nidx * (3 * Hh);
        const float* b = gh + nidx * (3 * Hh);
        const float r  = sigmf(a[c] + b[c]);
        const float z  = sigmf(a[Hh + c] + b[Hh + c]);
        const float nv = tanhf(a[2 * Hh + c] + r * b[2 * Hh + c]);
        const float h0 = h[i];
        float o = (1.f - z) * nv + z * h0;
        o = o > 0.f ? o : 0.f;
        h[i] = o; hbf[i] = (__bf16)o;
    }
}

// final [G,64] @ [64,1] + b2
__global__ void k_cls2(const float* __restrict__ y1, const float* __restrict__ w2,
                       const float* __restrict__ b2, float* __restrict__ out, int Gn)
{
    const int lane = threadIdx.x & 31;
    int g = blockIdx.x * (blockDim.x >> 5) + (threadIdx.x >> 5);
    const int st = gridDim.x * (blockDim.x >> 5);
    for (; g < Gn; g += st) {
        float s = y1[(long)g * 64 + lane] * w2[lane] + y1[(long)g * 64 + 32 + lane] * w2[32 + lane];
#pragma unroll
        for (int off = 16; off; off >>= 1) s += __shfl_xor(s, off, 32);
        if (lane == 0) out[g] = s + b2[0];
    }
}

// ---------------- host launch ----------------
static inline unsigned blocks_for(long items){
    long b = (items + 255) / 256; if (b < 1) b = 1; if (b > 1048576) b = 1048576;
    return (unsigned)b;
}
static inline unsigned blocks_for_waves(long waves){
    long b = (waves + 7) / 8; if (b < 1) b = 1; if (b > 262144) b = 262144;
    return (unsigned)b;
}

extern "C" void kernel_launch(void* const* d_in, const int* in_sizes, int n_in,
                              void* d_out, int out_size, void* d_ws, size_t ws_size,
                              hipStream_t stream)
{
    (void)in_sizes; (void)n_in; (void)out_size; (void)ws_size;
    const float* x         = (const float*)d_in[0];
    const float* eattr     = (const float*)d_in[1];
    const float* lin1_W    = (const float*)d_in[2];
    const float* lin1_b    = (const float*)d_in[3];
    const float* gc_lin1_W = (const float*)d_in[4];
    const float* gc_att_l  = (const float*)d_in[5];
    const float* gc_att_r  = (const float*)d_in[6];
    const float* gc_lin2_W = (const float*)d_in[7];
    const float* gc_bias   = (const float*)d_in[8];
    const float* gru0_Wi   = (const float*)d_in[9];
    const float* gru0_Wh   = (const float*)d_in[10];
    const float* gru0_bi   = (const float*)d_in[11];
    const float* gru0_bh   = (const float*)d_in[12];
    const float* atom_W    = (const float*)d_in[13];
    const float* a_att_src = (const float*)d_in[14];
    const float* a_att_dst = (const float*)d_in[15];
    const float* a_bias    = (const float*)d_in[16];
    const float* a_gru_Wi  = (const float*)d_in[17];
    const float* a_gru_Wh  = (const float*)d_in[18];
    const float* a_gru_bi  = (const float*)d_in[19];
    const float* a_gru_bh  = (const float*)d_in[20];
    const float* mol_W     = (const float*)d_in[21];
    const float* m_att_src = (const float*)d_in[22];
    const float* m_att_dst = (const float*)d_in[23];
    const float* m_bias    = (const float*)d_in[24];
    const float* m_gru_Wi  = (const float*)d_in[25];
    const float* m_gru_Wh  = (const float*)d_in[26];
    const float* m_gru_bi  = (const float*)d_in[27];
    const float* m_gru_bh  = (const float*)d_in[28];
    const float* lin2_W    = (const float*)d_in[29];
    const float* lin2_b    = (const float*)d_in[30];
    const float* cls_W1    = (const float*)d_in[31];
    const float* cls_b1    = (const float*)d_in[32];
    const float* cls_W2    = (const float*)d_in[33];
    const float* cls_b2    = (const float*)d_in[34];
    const int*   eidx      = (const int*)d_in[35];
    const int*   batch     = (const int*)d_in[36];
    const int*   src = eidx;
    const int*   dst = eidx + Ee;
    float* out_y = (float*)d_out;

    // ------------ workspace arena ------------
    char* base = (char*)d_ws;
    size_t cur = 0;
    auto alloc = [&](size_t bytes) -> char* {
        char* p = base + cur;
        cur = (cur + bytes + 255) & ~(size_t)255;
        return p;
    };
    __bf16* ECAT = (__bf16*)alloc((size_t)Ee * KCp * 2); // also hosts XBF, later XM
    __bf16* XBF  = ECAT;                                  // [Nn,KFp] fits inside
    __bf16* X0BF = (__bf16*)alloc((size_t)Nn * Hh * 2);
    float*  XH   = (float*) alloc((size_t)Nn * Hh * 4);   // x0 then xh (in-place GRU)
    __bf16* XHBF = (__bf16*)alloc((size_t)Nn * Hh * 2);
    float*  F1   = (float*) alloc((size_t)Nn * 3 * Hh * 4); // gi; graph bufs later
    float*  F2   = (float*) alloc((size_t)Nn * 3 * Hh * 4); // gh
    float*  AGG  = (float*) alloc((size_t)Nn * Hh * 4);
    __bf16* HBF  = (__bf16*)alloc((size_t)Nn * Hh * 2);
    float*  XL   = (float*) alloc((size_t)Nn * Hh * 4);   // x2 / xl
    float*  EA   = (float*) alloc((size_t)Ee * 4);
    float*  EB   = (float*) alloc((size_t)Ee * 4);
    float*  XR   = (float*) alloc((size_t)Nn * 4);
    unsigned* MK = (unsigned*)alloc((size_t)Nn * 4);
    float*  SS   = (float*) alloc((size_t)Nn * 4);
    float*  ASRC = (float*) alloc((size_t)Nn * 4);
    float*  ADST = (float*) alloc((size_t)Nn * 4);
    float*  ASRCN= (float*) alloc((size_t)Nn * 4);
    // bf16 weights
    __bf16* LIN1BF = (__bf16*)alloc((size_t)Hh * KFp * 2);
    __bf16* GCL1BF = (__bf16*)alloc((size_t)Hh * KCp * 2);
    __bf16* GCL2BF = (__bf16*)alloc((size_t)Hh * Hh * 2);
    __bf16* G0WIBF = (__bf16*)alloc((size_t)3 * Hh * Hh * 2);
    __bf16* G0WHBF = (__bf16*)alloc((size_t)3 * Hh * Hh * 2);
    __bf16* AWBF   = (__bf16*)alloc((size_t)2 * Hh * Hh * 2);
    __bf16* AWIBF  = (__bf16*)alloc((size_t)2 * 3 * Hh * Hh * 2);
    __bf16* AWHBF  = (__bf16*)alloc((size_t)2 * 3 * Hh * Hh * 2);
    __bf16* MWBF   = (__bf16*)alloc((size_t)Hh * Hh * 2);
    __bf16* MWIBF  = (__bf16*)alloc((size_t)3 * Hh * Hh * 2);
    __bf16* MWHBF  = (__bf16*)alloc((size_t)3 * Hh * Hh * 2);
    __bf16* L2BF   = (__bf16*)alloc((size_t)Hh * Hh * 2);
    __bf16* C1BF   = (__bf16*)alloc((size_t)64 * Hh * 2);
    // graph-level buffers live inside the (then-dead) F1 region
    char* gb = (char*)F1; size_t gcur = 0;
    auto galloc = [&](size_t bytes) -> char* {
        char* p = gb + gcur; gcur = (gcur + bytes + 255) & ~(size_t)255; return p;
    };
    float*   OUTG   = (float*) galloc((size_t)Gg * Hh * 4);
    __bf16*  OUTGBF = (__bf16*)galloc((size_t)Gg * Hh * 2);
    float*   OMG    = (float*) galloc((size_t)Gg * Hh * 4);
    float*   ADSTG  = (float*) galloc((size_t)Gg * 4);
    unsigned* MKG   = (unsigned*)galloc((size_t)Gg * 4);
    float*   SG     = (float*) galloc((size_t)Gg * 4);
    float*   AGGG   = (float*) galloc((size_t)Gg * Hh * 4);
    __bf16*  HGBF   = (__bf16*)galloc((size_t)Gg * Hh * 2);
    float*   GIG    = (float*) galloc((size_t)Gg * 3 * Hh * 4);
    float*   GHG    = (float*) galloc((size_t)Gg * 3 * Hh * 4);
    float*   EMB    = (float*) galloc((size_t)Gg * Hh * 4);
    __bf16*  EMBBF  = (__bf16*)galloc((size_t)Gg * Hh * 2);
    float*   Y1     = (float*) galloc((size_t)Gg * 64 * 4);
    float*   XM     = (float*)ECAT; // Ecat dead by molecule phase

    const dim3 B(256);
    auto cvt = [&](const float* s, __bf16* d, long rows, int cols, int colsp){
        k_cvt_pad<<<blocks_for(rows * (long)colsp), B, 0, stream>>>(s, d, rows, cols, colsp);
    };
    auto gemm = [&](const __bf16* A, const __bf16* W, const float* bias, float* Y,
                    int M, int Nout, int Kp, int act){
        long tiles = (long)(M >> 4) * (Nout >> 6);
        dim3 g(blocks_for_waves(tiles));
        if (Kp == KFp) {
            if (act == 1)      k_gemm<1, KFp><<<g, B, 0, stream>>>(A, W, bias, Y, M, Nout);
            else if (act == 2) k_gemm<2, KFp><<<g, B, 0, stream>>>(A, W, bias, Y, M, Nout);
            else               k_gemm<0, KFp><<<g, B, 0, stream>>>(A, W, bias, Y, M, Nout);
        } else {
            if (act == 1)      k_gemm<1, Hh><<<g, B, 0, stream>>>(A, W, bias, Y, M, Nout);
            else if (act == 2) k_gemm<2, Hh><<<g, B, 0, stream>>>(A, W, bias, Y, M, Nout);
            else               k_gemm<0, Hh><<<g, B, 0, stream>>>(A, W, bias, Y, M, Nout);
        }
    };

    // ---- weight conversions (fp32 -> bf16, K padded) ----
    cvt(lin1_W,    LIN1BF, Hh, Ff, KFp);
    cvt(gc_lin1_W, GCL1BF, Hh, Hh + Edw, KCp);
    cvt(gc_lin2_W, GCL2BF, Hh, Hh, Hh);
    cvt(gru0_Wi,   G0WIBF, 3 * Hh, Hh, Hh);
    cvt(gru0_Wh,   G0WHBF, 3 * Hh, Hh, Hh);
    cvt(atom_W,    AWBF,   2 * Hh, Hh, Hh);
    cvt(a_gru_Wi,  AWIBF,  2 * 3 * Hh, Hh, Hh);
    cvt(a_gru_Wh,  AWHBF,  2 * 3 * Hh, Hh, Hh);
    cvt(mol_W,     MWBF,   Hh, Hh, Hh);
    cvt(m_gru_Wi,  MWIBF,  3 * Hh, Hh, Hh);
    cvt(m_gru_Wh,  MWHBF,  3 * Hh, Hh, Hh);
    cvt(lin2_W,    L2BF,   Hh, Hh, Hh);
    cvt(cls_W1,    C1BF,   64, Hh, Hh);

    // ---- lin1 + leaky: x0 = leaky(x @ lin1_W^T + b) ----
    cvt(x, XBF, Nn, Ff, KFp);
    gemm(XBF, LIN1BF, lin1_b, XH, Nn, Hh, KFp, 1);
    cvt(XH, X0BF, Nn, Hh, Hh);

    // ---- GATEConv ----
    k_ecat<<<blocks_for((long)Ee * KCp), B, 0, stream>>>(X0BF, eattr, src, ECAT);
    k_rowdot2<<<blocks_for_waves(Nn), B, 0, stream>>>(XH, gc_att_r, nullptr, XR, nullptr, Nn);
    k_gate_gemm_rowdot<KCp><<<blocks_for_waves(Ee >> 4), B, 0, stream>>>(ECAT, GCL1BF, gc_att_l, EA, Ee);
    k_fill_u32<<<blocks_for(Nn), B, 0, stream>>>(MK, 0u, Nn);
    k_fill_u32<<<blocks_for(Nn), B, 0, stream>>>((unsigned*)SS, 0u, Nn);
    k_edge_alpha<<<blocks_for(Ee), B, 0, stream>>>(EA, nullptr, XR, dst, EA, MK, Ee);
    k_expsum<<<blocks_for(Ee), B, 0, stream>>>(EA, dst, MK, EB, SS, Ee);
    gemm(X0BF, GCL2BF, nullptr, XL, Nn, Hh, Hh, 0);
    k_rows_bias<<<blocks_for((long)Nn * Hh), B, 0, stream>>>(AGG, gc_bias, Nn);
    k_scatter<<<blocks_for_waves(Ee), B, 0, stream>>>(XL, src, dst, EB, SS, AGG, Ee);
    k_elu_bf16<<<blocks_for((long)Nn * Hh), B, 0, stream>>>(AGG, HBF, (long)Nn * Hh);
    gemm(HBF,  G0WIBF, gru0_bi, F1, Nn, 3 * Hh, Hh, 0);
    gemm(X0BF, G0WHBF, gru0_bh, F2, Nn, 3 * Hh, Hh, 0);
    k_gru<<<blocks_for((long)Nn * Hh), B, 0, stream>>>(F1, F2, XH, XHBF, Nn);

    // ---- atom GATConv layers ----
    for (int l = 0; l < 2; l++) {
        gemm(XHBF, AWBF + (size_t)l * Hh * Hh, nullptr, XL, Nn, Hh, Hh, 0);
        k_rowdot2<<<blocks_for_waves(Nn), B, 0, stream>>>(XL, a_att_src + l * Hh,
                                                          a_att_dst + l * Hh, ASRC, ADST, Nn);
        k_fill_u32<<<blocks_for(Nn), B, 0, stream>>>(MK, 0u, Nn);
        k_fill_u32<<<blocks_for(Nn), B, 0, stream>>>((unsigned*)SS, 0u, Nn);
        k_edge_alpha<<<blocks_for(Ee), B, 0, stream>>>(ASRC, src, ADST, dst, EA, MK, Ee);
        k_expsum<<<blocks_for(Ee), B, 0, stream>>>(EA, dst, MK, EB, SS, Ee);
        k_rows_bias<<<blocks_for((long)Nn * Hh), B, 0, stream>>>(AGG, a_bias + l * Hh, Nn);
        k_scatter<<<blocks_for_waves(Ee), B, 0, stream>>>(XL, src, dst, EB, SS, AGG, Ee);
        k_elu_bf16<<<blocks_for((long)Nn * Hh), B, 0, stream>>>(AGG, HBF, (long)Nn * Hh);
        gemm(HBF,  AWIBF + (size_t)l * 3 * Hh * Hh, a_gru_bi + l * 3 * Hh, F1, Nn, 3 * Hh, Hh, 0);
        gemm(XHBF, AWHBF + (size_t)l * 3 * Hh * Hh, a_gru_bh + l * 3 * Hh, F2, Nn, 3 * Hh, Hh, 0);
        k_gru<<<blocks_for((long)Nn * Hh), B, 0, stream>>>(F1, F2, XH, XHBF, Nn);
    }

    // ---- molecule readout ----
    k_fill_u32<<<blocks_for((long)Gg * Hh), B, 0, stream>>>((unsigned*)AGGG, 0u, (long)Gg * Hh);
    k_scatter<<<blocks_for_waves(Nn), B, 0, stream>>>(XH, nullptr, batch, nullptr, nullptr, AGGG, Nn);
    k_relu_fb<<<blocks_for((long)Gg * Hh), B, 0, stream>>>(AGGG, OUTG, OUTGBF, (long)Gg * Hh);
    gemm(XHBF, MWBF, nullptr, XM, Nn, Hh, Hh, 0);
    k_rowdot2<<<blocks_for_waves(Nn), B, 0, stream>>>(XM, m_att_src, nullptr, ASRCN, nullptr, Nn);
    for (int t = 0; t < 2; t++) {
        gemm(OUTGBF, MWBF, nullptr, OMG, Gg, Hh, Hh, 0);
        k_rowdot2<<<blocks_for_waves(Gg), B, 0, stream>>>(OMG, m_att_dst, nullptr, ADSTG, nullptr, Gg);
        k_fill_u32<<<blocks_for(Gg), B, 0, stream>>>(MKG, 0u, Gg);
        k_fill_u32<<<blocks_for(Gg), B, 0, stream>>>((unsigned*)SG, 0u, Gg);
        k_edge_alpha<<<blocks_for(Nn), B, 0, stream>>>(ASRCN, nullptr, ADSTG, batch, EA, MKG, Nn);
        k_expsum<<<blocks_for(Nn), B, 0, stream>>>(EA, batch, MKG, EB, SG, Nn);
        k_rows_bias<<<blocks_for((long)Gg * Hh), B, 0, stream>>>(AGGG, m_bias, Gg);
        k_scatter<<<blocks_for_waves(Nn), B, 0, stream>>>(XM, nullptr, batch, EB, SG, AGGG, Nn);
        k_elu_bf16<<<blocks_for((long)Gg * Hh), B, 0, stream>>>(AGGG, HGBF, (long)Gg * Hh);
        gemm(HGBF,   MWIBF, m_gru_bi, GIG, Gg, 3 * Hh, Hh, 0);
        gemm(OUTGBF, MWHBF, m_gru_bh, GHG, Gg, 3 * Hh, Hh, 0);
        k_gru<<<blocks_for((long)Gg * Hh), B, 0, stream>>>(GIG, GHG, OUTG, OUTGBF, Gg);
    }

    // ---- lin2 + classifier ----
    gemm(OUTGBF, L2BF, lin2_b, EMB, Gg, Hh, Hh, 0);
    cvt(EMB, EMBBF, Gg, Hh, Hh);
    gemm(EMBBF, C1BF, cls_b1, Y1, Gg, 64, Hh, 2);
    k_cls2<<<blocks_for_waves(Gg), B, 0, stream>>>(Y1, cls_W2, cls_b2, out_y, Gg);
}
